// DiscriminatorSTFT_78649441124711
// MI455X (gfx1250) — compile-verified, matrix operand order checked by reading, not run on hardware
//
#include <hip/hip_runtime.h>
#include <math.h>
#include <stdint.h>

// ---------------- types for WMMA ----------------
typedef __attribute__((ext_vector_type(16))) _Float16 v16h;
typedef __attribute__((ext_vector_type(8)))  float    v8f;

union AFrag {
    v16h     v;
    uint32_t u[8];
    _Float16 h[16];
};

#define NFFT     1024
#define HOP      512
#define NBINS    513
#define NFRAMES  256
#define BATCH    16
#define TLEN     131072
#define NROWS    (BATCH * NFRAMES)   /* 4096 rows = (b,h) or (b,frame) */
#define NCOLP    1040                /* 2*513 = 1026 padded to 16-multiple */
#define CH       32
#define NEG_SLOPE 0.2f

#define WMMA16(a, b, c) __builtin_amdgcn_wmma_f32_16x16x32_f16(false, (a), false, (b), (short)0, (c), false, false)

// ============ 1. windowed frames (f32 -> f16) ============
__global__ void k_frames(const float* __restrict__ x, _Float16* __restrict__ frames) {
    int id = blockIdx.x * blockDim.x + threadIdx.x;       // < 4096*1024
    int n  = id & (NFFT - 1);
    int fr = id >> 10;
    int b  = fr >> 8;
    int f  = fr & 255;
    long t = (long)f * HOP + n;
    float v = (t < TLEN) ? x[(long)b * TLEN + t] : 0.0f;
    float win = 0.5f * (1.0f - __cosf(6.28318530717958647692f * (float)n / (float)NFFT));
    frames[id] = (_Float16)(v * win);
}

// ============ 2. transposed DFT basis [NCOLP][NFFT] f16 ============
// basisT[col][n]: col even -> cos(2*pi*n*(col/2)/N), col odd -> -sin(...), col>=1026 -> 0
__global__ void k_basisT(_Float16* __restrict__ basisT) {
    int id = blockIdx.x * blockDim.x + threadIdx.x;
    if (id >= NCOLP * NFFT) return;
    int n   = id & (NFFT - 1);
    int col = id >> 10;
    float val = 0.0f;
    if (col < 2 * NBINS) {
        int k  = col >> 1;
        int ph = (n * k) & (NFFT - 1);          // exact modular phase
        float th = 6.28318530717958647692f * (float)ph / (float)NFFT;
        val = (col & 1) ? -sinf(th) : cosf(th);
    }
    basisT[id] = (_Float16)val;
}

// ============ 3. DFT GEMM: C[4096][1040] = frames[4096][1024] x basis ============
// 2 M-tiles per wave: B fragment loaded once, 2 WMMAs per K-chunk.
__global__ void k_dft_gemm(const _Float16* __restrict__ A,
                           const _Float16* __restrict__ BT,
                           float* __restrict__ C) {
    int lane = threadIdx.x & 31;
    int wave = (blockIdx.x * blockDim.x + threadIdx.x) >> 5;
    const int NT = NCOLP / 16;                  // 65 n-tiles
    int mtp = wave / NT;                        // 0..127 (pairs of m-tiles)
    int nt  = wave % NT;
    if (mtp >= NROWS / 32) return;
    int m    = lane & 15;
    int hi8  = (lane & 16) ? 8 : 0;
    int row0 = mtp * 32 + m;
    int row1 = row0 + 16;
    int col  = nt * 16 + m;                     // B: lane -> N column
    v8f acc0 = {}, acc1 = {};
    const uint32_t* ar0 = (const uint32_t*)(A + (long)row0 * NFFT);
    const uint32_t* ar1 = (const uint32_t*)(A + (long)row1 * NFFT);
    const uint32_t* bt  = (const uint32_t*)(BT + (long)col * NFFT);
    for (int kc = 0; kc < NFFT / 32; ++kc) {
        AFrag a0, a1, b;
        #pragma unroll
        for (int r = 0; r < 8; ++r) {
            int kb = ((r & 4) << 2) + hi8 + ((r & 3) << 1);  // ISA 16-bit A/B K layout
            int ki = (kc * 32 + kb) >> 1;
            a0.u[r] = ar0[ki];
            a1.u[r] = ar1[ki];
            b.u[r]  = bt[ki];
        }
        acc0 = WMMA16(a0.v, b.v, acc0);
        acc1 = WMMA16(a1.v, b.v, acc1);
    }
    #pragma unroll
    for (int i = 0; i < 8; ++i) {
        int cc  = nt * 16 + (lane & 15);
        int rr0 = mtp * 32 + i + hi8;
        C[(long)rr0 * NCOLP + cc]        = acc0[i];
        C[(long)(rr0 + 16) * NCOLP + cc] = acc1[i];
    }
}

// ============ 4. magnitude + per-frame max normalize -> spec f16 ============
__global__ void k_magnorm(const float* __restrict__ C, _Float16* __restrict__ spec) {
    __shared__ float mags[NBINS];
    __shared__ float red[256];
    int row = blockIdx.x;
    int tid = threadIdx.x;
    const float* cr = C + (long)row * NCOLP;
    float mx = 0.0f;
    for (int j = tid; j < NBINS; j += 256) {
        float re = cr[2*j], im = cr[2*j + 1];
        float mg = sqrtf(re * re + im * im);
        mags[j] = mg;
        mx = fmaxf(mx, mg);
    }
    red[tid] = mx;
    __syncthreads();
    for (int s = 128; s > 0; s >>= 1) {
        if (tid < s) red[tid] = fmaxf(red[tid], red[tid + s]);
        __syncthreads();
    }
    float inv = 1.0f / red[0];
    for (int j = tid; j < NBINS; j += 256)
        spec[(long)row * NBINS + j] = (_Float16)(mags[j] * inv);
}

// ============ 5. weight repacks (transpose to cout-major for b128 B loads) ============
// w0 (3,9,2,32): cin=0 slice -> w0fT[cout(32)][tap(32)], taps 27..31 zero
__global__ void k_w0_repackT(const float* __restrict__ w0, _Float16* __restrict__ w0fT) {
    int id = threadIdx.x;            // one block of 1024
    int cout = id >> 5, tap = id & 31;
    float v = (tap < 27) ? w0[tap * 64 + cout] : 0.0f;   // ((tap)*2 + 0)*32 + cout
    w0fT[id] = (_Float16)v;
}
// w (KH,KW,32,32) f32 [tap][cin][cout] -> f16 [tap][cout][cin]
__global__ void k_wrepackT(const float* __restrict__ src, _Float16* __restrict__ dst, int ntaps) {
    int id = blockIdx.x * blockDim.x + threadIdx.x;
    if (id >= ntaps * 1024) return;
    int tap = id >> 10;
    int cout = (id >> 5) & 31;
    int cin  = id & 31;
    dst[id] = (_Float16)src[((long)tap * 32 + cin) * 32 + cout];
}

// ============ 6. conv0: 1-channel spec, K=27 taps in one WMMA chunk ============
__global__ void k_conv0(const _Float16* __restrict__ spec, const _Float16* __restrict__ w0fT,
                        const float* __restrict__ bias,
                        float* __restrict__ fmap, _Float16* __restrict__ act) {
    int lane = threadIdx.x & 31;
    int wave = (blockIdx.x * blockDim.x + threadIdx.x) >> 5;
    const int WT = 33;                              // ceil(513/16)
    int row = wave / WT;
    int wt  = wave % WT;
    if (row >= NROWS) return;
    int b = row >> 8, h = row & 255;
    int m    = lane & 15;
    int hi8  = (lane & 16) ? 8 : 0;
    int w    = wt * 16 + m;
    int c0   = lane & 15;
    const uint32_t* q0 = (const uint32_t*)(w0fT + (long)c0 * 32);
    const uint32_t* q1 = (const uint32_t*)(w0fT + (long)(c0 + 16) * 32);
    AFrag a, b0, b1;
    #pragma unroll
    for (int r = 0; r < 8; ++r) {
        int kb = ((r & 4) << 2) + hi8 + ((r & 3) << 1);
        #pragma unroll
        for (int e = 0; e < 2; ++e) {
            int kk = kb + e;
            _Float16 av = (_Float16)0.0f;
            if (kk < 27 && w < NBINS) {
                int kh = kk / 9, kw = kk - kh * 9;
                int hin = h + kh - 1;
                int win = w + kw - 4;
                if ((unsigned)hin < 256u && (unsigned)win < (unsigned)NBINS)
                    av = spec[((long)(b * 256 + hin)) * NBINS + win];
            }
            a.h[2*r + e] = av;
        }
        b0.u[r] = q0[kb >> 1];
        b1.u[r] = q1[kb >> 1];
    }
    v8f acc0 = {}, acc1 = {};
    acc0 = WMMA16(a.v, b0.v, acc0);
    acc1 = WMMA16(a.v, b1.v, acc1);
    #pragma unroll
    for (int i = 0; i < 8; ++i) {
        int wo = wt * 16 + i + hi8;
        if (wo < NBINS) {
            long base = (((long)row) * NBINS + wo) * CH;
            float v0 = acc0[i] + bias[c0];
            v0 = v0 >= 0.0f ? v0 : NEG_SLOPE * v0;
            fmap[base + c0] = v0;  act[base + c0] = (_Float16)v0;
            float v1 = acc1[i] + bias[c0 + 16];
            v1 = v1 >= 0.0f ? v1 : NEG_SLOPE * v1;
            fmap[base + c0 + 16] = v1;  act[base + c0 + 16] = (_Float16)v1;
        }
    }
}

// ============ 7. generic 32->32 conv: 2 M-tiles/wave, 4 WMMA per tap ============
__global__ void k_conv(const _Float16* __restrict__ in, int Win,
                       float* __restrict__ fmap, _Float16* __restrict__ act, int Wout,
                       const _Float16* __restrict__ wtT, const float* __restrict__ bias,
                       int KW, int sw, int pw, int dh) {
    int lane = threadIdx.x & 31;
    int wave = (blockIdx.x * blockDim.x + threadIdx.x) >> 5;
    int WT  = (Wout + 15) >> 4;
    int WTp = (WT + 1) >> 1;                 // pairs of w-tiles
    int row = wave / WTp;
    int pt  = wave % WTp;
    if (row >= NROWS) return;
    int b = row >> 8, h = row & 255;
    int m    = lane & 15;
    int hi8  = (lane & 16) ? 8 : 0;
    int wo0  = pt * 32 + m;
    int wo1  = wo0 + 16;
    int c0   = lane & 15;
    v8f acc00 = {}, acc01 = {}, acc10 = {}, acc11 = {};
    for (int kh = 0; kh < 3; ++kh) {
        int hin = h + (kh - 1) * dh;
        if ((unsigned)hin >= 256u) continue;              // whole tap row is zero-pad
        const _Float16* inrow = in + ((long)(b * 256 + hin)) * Win * CH;
        for (int kw = 0; kw < KW; ++kw) {
            int tap = kh * KW + kw;
            int wi0 = wo0 * sw - pw + kw;
            int wi1 = wi0 + 16 * sw;
            bool val0 = (wo0 < Wout) && ((unsigned)wi0 < (unsigned)Win);
            bool val1 = (wo1 < Wout) && ((unsigned)wi1 < (unsigned)Win);
            const uint32_t* p0 = (const uint32_t*)(inrow + (long)wi0 * CH);
            const uint32_t* p1 = (const uint32_t*)(inrow + (long)wi1 * CH);
            const uint32_t* q0 = (const uint32_t*)(wtT + ((long)tap * 32 + c0) * 32);
            const uint32_t* q1 = (const uint32_t*)(wtT + ((long)tap * 32 + c0 + 16) * 32);
            AFrag a0, a1, b0, b1;
            #pragma unroll
            for (int r = 0; r < 8; ++r) {
                int kb = ((r & 4) << 2) + hi8 + ((r & 3) << 1);
                int ki = kb >> 1;
                a0.u[r] = val0 ? p0[ki] : 0u;             // contiguous channel pairs
                a1.u[r] = val1 ? p1[ki] : 0u;
                b0.u[r] = q0[ki];                         // cout-major weights: b128s
                b1.u[r] = q1[ki];
            }
            acc00 = WMMA16(a0.v, b0.v, acc00);
            acc01 = WMMA16(a0.v, b1.v, acc01);
            acc10 = WMMA16(a1.v, b0.v, acc10);
            acc11 = WMMA16(a1.v, b1.v, acc11);
        }
    }
    #pragma unroll
    for (int i = 0; i < 8; ++i) {
        int mm = i + hi8;
        int wout0 = pt * 32 + mm;
        int wout1 = wout0 + 16;
        if (wout0 < Wout) {
            long base = (((long)row) * Wout + wout0) * CH;
            float v0 = acc00[i] + bias[c0];
            v0 = v0 >= 0.0f ? v0 : NEG_SLOPE * v0;
            fmap[base + c0] = v0;  act[base + c0] = (_Float16)v0;
            float v1 = acc01[i] + bias[c0 + 16];
            v1 = v1 >= 0.0f ? v1 : NEG_SLOPE * v1;
            fmap[base + c0 + 16] = v1;  act[base + c0 + 16] = (_Float16)v1;
        }
        if (wout1 < Wout) {
            long base = (((long)row) * Wout + wout1) * CH;
            float v0 = acc10[i] + bias[c0];
            v0 = v0 >= 0.0f ? v0 : NEG_SLOPE * v0;
            fmap[base + c0] = v0;  act[base + c0] = (_Float16)v0;
            float v1 = acc11[i] + bias[c0 + 16];
            v1 = v1 >= 0.0f ? v1 : NEG_SLOPE * v1;
            fmap[base + c0 + 16] = v1;  act[base + c0 + 16] = (_Float16)v1;
        }
    }
}

// ============ 8. projection head: 32 -> 1 channel, 3x3 (tiny, VALU) ============
__global__ void k_convp(const _Float16* __restrict__ act4, const float* __restrict__ wp,
                        const float* __restrict__ bp, float* __restrict__ out) {
    int id = blockIdx.x * blockDim.x + threadIdx.x;
    if (id >= NROWS * 65) return;
    int wo  = id % 65;
    int row = id / 65;
    int b = row >> 8, h = row & 255;
    float accv = bp[0];
    for (int kh = 0; kh < 3; ++kh) {
        int hin = h + kh - 1;
        if ((unsigned)hin >= 256u) continue;
        for (int kw = 0; kw < 3; ++kw) {
            int wi = wo + kw - 1;
            if ((unsigned)wi >= 65u) continue;
            const _Float16* p = act4 + (((long)(b * 256 + hin)) * 65 + wi) * CH;
            const float* wv = wp + (kh * 3 + kw) * CH;    // (3,3,32,1)
            #pragma unroll
            for (int c = 0; c < CH; ++c) accv += (float)p[c] * wv[c];
        }
    }
    out[id] = accv;
}

// =================================================================
extern "C" void kernel_launch(void* const* d_in, const int* in_sizes, int n_in,
                              void* d_out, int out_size, void* d_ws, size_t ws_size,
                              hipStream_t stream) {
    (void)in_sizes; (void)n_in; (void)out_size; (void)ws_size;
    const float* x  = (const float*)d_in[0];
    const float* w0 = (const float*)d_in[1];   const float* b0 = (const float*)d_in[2];
    const float* w1 = (const float*)d_in[3];   const float* b1 = (const float*)d_in[4];
    const float* w2 = (const float*)d_in[5];   const float* b2 = (const float*)d_in[6];
    const float* w3 = (const float*)d_in[7];   const float* b3 = (const float*)d_in[8];
    const float* w4 = (const float*)d_in[9];   const float* b4 = (const float*)d_in[10];
    const float* wp = (const float*)d_in[11];  const float* bp = (const float*)d_in[12];

    // ---- d_out layout: (z, fmap0..fmap4) flat ----
    float* outp  = (float*)d_out;
    float* zout  = outp;                               // 4096*65*1
    float* fmap0 = zout  + (long)NROWS * 65;           // 4096*513*32
    float* fmap1 = fmap0 + (long)NROWS * 513 * CH;     // 4096*257*32
    float* fmap2 = fmap1 + (long)NROWS * 257 * CH;     // 4096*129*32
    float* fmap3 = fmap2 + (long)NROWS * 129 * CH;     // 4096*65*32
    float* fmap4 = fmap3 + (long)NROWS * 65  * CH;     // 4096*65*32

    // ---- workspace carve-up ----
    char* ws = (char*)d_ws;
    size_t off = 0;
    auto carve = [&](size_t bytes) { void* p = ws + off; off = (off + bytes + 255) & ~(size_t)255; return p; };
    _Float16* frames = (_Float16*)carve((size_t)NROWS * NFFT * 2);
    _Float16* basisT = (_Float16*)carve((size_t)NCOLP * NFFT * 2);
    float*    dftC   = (float*)   carve((size_t)NROWS * NCOLP * 4);
    _Float16* spec   = (_Float16*)carve((size_t)NROWS * NBINS * 2);
    _Float16* act0   = (_Float16*)carve((size_t)NROWS * 513 * CH * 2);
    _Float16* act1   = (_Float16*)carve((size_t)NROWS * 257 * CH * 2);
    _Float16* act2   = (_Float16*)carve((size_t)NROWS * 129 * CH * 2);
    _Float16* act3   = (_Float16*)carve((size_t)NROWS * 65  * CH * 2);
    _Float16* act4   = (_Float16*)carve((size_t)NROWS * 65  * CH * 2);
    _Float16* w0fT   = (_Float16*)carve(32 * 32 * 2);
    _Float16* w1fT   = (_Float16*)carve(27648 * 2);
    _Float16* w2fT   = (_Float16*)carve(27648 * 2);
    _Float16* w3fT   = (_Float16*)carve(27648 * 2);
    _Float16* w4fT   = (_Float16*)carve(9216 * 2);

    // ---- STFT pipeline ----
    k_frames <<<(NROWS * NFFT) / 256, 256, 0, stream>>>(x, frames);
    k_basisT <<<(NCOLP * NFFT + 255) / 256, 256, 0, stream>>>(basisT);
    k_dft_gemm<<<((NROWS / 32) * (NCOLP / 16)) / 8, 256, 0, stream>>>(frames, basisT, dftC);
    k_magnorm<<<NROWS, 256, 0, stream>>>(dftC, spec);

    // ---- weight repacks ----
    k_w0_repackT<<<1, 1024, 0, stream>>>(w0, w0fT);
    k_wrepackT<<<(27 * 1024 + 255) / 256, 256, 0, stream>>>(w1, w1fT, 27);
    k_wrepackT<<<(27 * 1024 + 255) / 256, 256, 0, stream>>>(w2, w2fT, 27);
    k_wrepackT<<<(27 * 1024 + 255) / 256, 256, 0, stream>>>(w3, w3fT, 27);
    k_wrepackT<<<( 9 * 1024 + 255) / 256, 256, 0, stream>>>(w4, w4fT, 9);

    // ---- convs (WMMA implicit GEMM) ----
    k_conv0<<<(NROWS * 33) / 8, 256, 0, stream>>>(spec, w0fT, b0, fmap0, act0);
    k_conv <<<(NROWS * 9) / 8, 256, 0, stream>>>(act0, 513, fmap1, act1, 257, w1fT, b1, 9, 2, 4, 1);
    k_conv <<<(NROWS * 5) / 8, 256, 0, stream>>>(act1, 257, fmap2, act2, 129, w2fT, b2, 9, 2, 4, 2);
    k_conv <<<(NROWS * 3) / 8, 256, 0, stream>>>(act2, 129, fmap3, act3,  65, w3fT, b3, 9, 2, 4, 4);
    k_conv <<<(NROWS * 3) / 8, 256, 0, stream>>>(act3,  65, fmap4, act4,  65, w4fT, b4, 3, 1, 1, 1);
    k_convp<<<(NROWS * 65 + 255) / 256, 256, 0, stream>>>(act4, wp, bp, zout);
}
